// AnimeGNN_40003325395140
// MI455X (gfx1250) — compile-verified, matrix operand order checked by reading, not run on hardware
//
#include <hip/hip_runtime.h>
#include <hip/hip_bf16.h>
#include <math.h>

// ---------------------------------------------------------------------------
// 2-layer GCN for MI455X (gfx1250, wave32).
//   GEMMs via v_wmma_f32_16x16x32_bf16 (fp32 accum).
//   Weights pre-packed once into transposed bf16 so each wave's B fragment is
//   two contiguous global_load_b128 (was 16 strided global_load_b32 + 8 cvt).
//   Edge scatter via float4 gathers + global_atomic_add_f32 (unsafeAtomicAdd).
// Whole working set (~100MB) fits in the 192MB L2, so the scatter phase runs
// at L2 rate; GEMM flops (~5 GF) are negligible once on the WMMA path.
// ---------------------------------------------------------------------------

typedef __attribute__((ext_vector_type(16))) __bf16 v16bf;
typedef __attribute__((ext_vector_type(8)))  __bf16 v8bf;
typedef __attribute__((ext_vector_type(8)))  float  v8f;

#define N_NODES 50000
#define FDIM    128     // hidden == output dim

// ---------------------------------------------------------------------------
// Degree / normalization kernels
// ---------------------------------------------------------------------------
__global__ void deg_init_k(unsigned* __restrict__ deg, int n) {
    int i = blockIdx.x * blockDim.x + threadIdx.x;
    if (i < n) deg[i] = 1u;                       // self-loop
}

__global__ void deg_accum_k(const int* __restrict__ dst, unsigned* __restrict__ deg, int E) {
    int e = blockIdx.x * blockDim.x + threadIdx.x;
    if (e < E) atomicAdd(&deg[dst[e]], 1u);       // global_atomic_add_u32
}

__global__ void dinv_k(const unsigned* __restrict__ deg, float* __restrict__ dinv, int n) {
    int i = blockIdx.x * blockDim.x + threadIdx.x;
    if (i < n) dinv[i] = rsqrtf((float)deg[i]);   // v_rsq_f32
}

// ---------------------------------------------------------------------------
// Pack W[K x 128] fp32 (row-major) -> Wt[128 x K] bf16 (transposed).
// Tiny one-shot kernel; makes GEMM B-fragment loads contiguous b128s.
// ---------------------------------------------------------------------------
template <int K>
__global__ void pack_w_bf16_k(const float* __restrict__ W, __bf16* __restrict__ Wt) {
    int i = blockIdx.x * blockDim.x + threadIdx.x;      // over K*128 elements
    if (i >= K * FDIM) return;
    int k = i / FDIM;
    int n = i & (FDIM - 1);
    Wt[(size_t)n * K + k] = (__bf16)W[i];
}

// ---------------------------------------------------------------------------
// GEMM: C[M x 128] = A[M x K] * B[K x 128], fp32 A/C, pre-packed bf16 Bt.
// One wave per 16x16 C tile; 8 waves per block -> block covers 16 x 128.
// Register layouts per CDNA5 ISA 7.12.2 (16-bit A 16x32, B 32x16):
//   lane L: m(or n) = L&15 ; K-half = (L>>4)*8 ; VGPR 0..3 -> K 0..15 pairs,
//   VGPR 4..7 -> K 16..31 pairs.
// ---------------------------------------------------------------------------
template <int K>
__global__ void gemm_wmma_bf16_k(const float* __restrict__ A,
                                 const __bf16* __restrict__ Bt,  // [128 x K] bf16
                                 float* __restrict__ C) {
    const int lane  = threadIdx.x & 31;
    const int wave  = threadIdx.x >> 5;
    const int mBase = blockIdx.x * 16;            // M = 50000 = 3125 * 16 exact
    const int nBase = wave * 16;
    const int mrow  = mBase + (lane & 15);
    const int ncol  = nBase + (lane & 15);
    const int kh    = (lane >> 4) * 8;            // per-half K sub-offset

    v8f acc = {};
#pragma unroll
    for (int k0 = 0; k0 < K; k0 += 32) {
        v16bf a, b;
        // ---- A tile 16x32 (row-major fp32 -> bf16 in-register) ----
        const float* arow = A + (size_t)mrow * K + k0 + kh;
#pragma unroll
        for (int v = 0; v < 4; ++v) {
            float2 f0 = *(const float2*)(arow + 2 * v);        // K = kh+2v,+1
            float2 f1 = *(const float2*)(arow + 16 + 2 * v);   // K = 16+kh+2v,+1
            a[2 * v]     = (__bf16)f0.x;  a[2 * v + 1]     = (__bf16)f0.y;
            a[8 + 2 * v] = (__bf16)f1.x;  a[8 + 2 * v + 1] = (__bf16)f1.y;
        }
        // ---- B tile 32x16: two contiguous 16B chunks from transposed bf16 ----
        const __bf16* bp = Bt + (size_t)ncol * K + k0 + kh;    // 16B aligned
        v8bf lo = *(const v8bf*)bp;          // K = kh .. kh+7
        v8bf hi = *(const v8bf*)(bp + 16);   // K = 16+kh .. 16+kh+7
#pragma unroll
        for (int i = 0; i < 8; ++i) { b[i] = lo[i]; b[8 + i] = hi[i]; }

        acc = __builtin_amdgcn_wmma_f32_16x16x32_bf16(
            /*neg_a=*/false, a, /*neg_b=*/false, b,
            /*c_mod=*/(short)0, acc, /*reuse_a=*/false, /*reuse_b=*/false);
    }
    // ---- store C: VGPR r holds row mBase + r + 8*(lane>>4), col = ncol ----
    float* crow = C + (size_t)(mBase + (lane >> 4) * 8) * FDIM + ncol;
#pragma unroll
    for (int r = 0; r < 8; ++r) crow[(size_t)r * FDIM] = acc[r];
}

// ---------------------------------------------------------------------------
// Aggregation: out[i,:] = H[i,:]*dinv[i]^2 (+bias)   (self-loop message init)
// ---------------------------------------------------------------------------
__global__ void agg_init_k(const float* __restrict__ H,
                           const float* __restrict__ dinv,
                           const float* __restrict__ bias,   // may be null
                           float* __restrict__ out, int total) {
    int i = blockIdx.x * blockDim.x + threadIdx.x;
    if (i >= total) return;
    int node = i >> 7;            // /128
    int j    = i & (FDIM - 1);
    float d  = dinv[node];
    float v  = H[i] * d * d;
    if (bias) v += bias[j];
    out[i] = v;
}

// One wave per edge; lane l covers features [4l, 4l+4).
__global__ void agg_edges_k(const int* __restrict__ src,
                            const int* __restrict__ dst,
                            const float* __restrict__ H,
                            const float* __restrict__ dinv,
                            float* __restrict__ out, int E) {
    int e = blockIdx.x * (blockDim.x >> 5) + (threadIdx.x >> 5);
    if (e >= E) return;
    int lane = threadIdx.x & 31;
    int s = src[e], d = dst[e];
    float w = dinv[s] * dinv[d];
    const float4 v = *(const float4*)(H + (size_t)s * FDIM + lane * 4);
    float* o = out + (size_t)d * FDIM + lane * 4;
    unsafeAtomicAdd(o + 0, v.x * w);   // global_atomic_add_f32
    unsafeAtomicAdd(o + 1, v.y * w);
    unsafeAtomicAdd(o + 2, v.z * w);
    unsafeAtomicAdd(o + 3, v.w * w);
}

// out = relu(out + b)
__global__ void bias_relu_k(float* __restrict__ a, const float* __restrict__ b, int total) {
    int i = blockIdx.x * blockDim.x + threadIdx.x;
    if (i < total) a[i] = fmaxf(a[i] + b[i & (FDIM - 1)], 0.0f);
}

// ---------------------------------------------------------------------------
// Host-side orchestration
// ---------------------------------------------------------------------------
static inline size_t alignup(size_t x) { return (x + 255) & ~(size_t)255; }

extern "C" void kernel_launch(void* const* d_in, const int* in_sizes, int n_in,
                              void* d_out, int out_size, void* d_ws, size_t ws_size,
                              hipStream_t stream) {
    const float* x    = (const float*)d_in[0];       // [N, 256]
    const int*   edge = (const int*)d_in[1];         // [2, E]
    const float* W1   = (const float*)d_in[2];       // [256, 128]
    const float* b1   = (const float*)d_in[3];       // [128]
    const float* W2   = (const float*)d_in[4];       // [128, 128]
    const float* b2   = (const float*)d_in[5];       // [128]
    float*       z    = (float*)d_out;               // [N, 128]

    const int E = in_sizes[1] / 2;
    const int* src = edge;
    const int* dst = edge + E;

    // workspace carve-up
    char* ws = (char*)d_ws;
    size_t off = 0;
    unsigned* deg  = (unsigned*)(ws + off); off += alignup((size_t)N_NODES * 4);
    float*    dinv = (float*)   (ws + off); off += alignup((size_t)N_NODES * 4);
    float*    H    = (float*)   (ws + off); off += alignup((size_t)N_NODES * FDIM * 4);
    float*    A1   = (float*)   (ws + off); off += alignup((size_t)N_NODES * FDIM * 4);
    __bf16*   Wt1  = (__bf16*)  (ws + off); off += alignup((size_t)256 * FDIM * 2);
    __bf16*   Wt2  = (__bf16*)  (ws + off); off += alignup((size_t)128 * FDIM * 2);
    (void)ws_size; (void)n_in; (void)out_size;

    const int total = N_NODES * FDIM;
    const dim3 blk(256);
    const int gNodes = (N_NODES + 255) / 256;
    const int gEdges = (E + 255) / 256;
    const int gElem  = (total + 255) / 256;
    const int gGemm  = N_NODES / 16;                 // 3125, exact
    const int gScat  = (E + 7) / 8;                  // 8 edges / 256-thread block

    // --- normalization + weight packing (independent, tiny) ---
    deg_init_k <<<gNodes, blk, 0, stream>>>(deg, N_NODES);
    deg_accum_k<<<gEdges, blk, 0, stream>>>(dst, deg, E);
    dinv_k     <<<gNodes, blk, 0, stream>>>(deg, dinv, N_NODES);
    pack_w_bf16_k<256><<<(256 * FDIM + 255) / 256, blk, 0, stream>>>(W1, Wt1);
    pack_w_bf16_k<128><<<(128 * FDIM + 255) / 256, blk, 0, stream>>>(W2, Wt2);

    // --- layer 1: H = x @ W1 ; A1 = scatter(H) ; A1 = relu(A1 + b1) ---
    gemm_wmma_bf16_k<256><<<gGemm, blk, 0, stream>>>(x, Wt1, H);
    agg_init_k <<<gElem, blk, 0, stream>>>(H, dinv, nullptr, A1, total);
    agg_edges_k<<<gScat, blk, 0, stream>>>(src, dst, H, dinv, A1, E);
    bias_relu_k<<<gElem, blk, 0, stream>>>(A1, b1, total);

    // --- layer 2: H = A1 @ W2 ; z = scatter(H) + b2 ---
    gemm_wmma_bf16_k<128><<<gGemm, blk, 0, stream>>>(A1, Wt2, H);
    agg_init_k <<<gElem, blk, 0, stream>>>(H, dinv, b2, z, total);
    agg_edges_k<<<gScat, blk, 0, stream>>>(src, dst, H, dinv, z, E);
}
